// HAN_31250182046569
// MI455X (gfx1250) — matrix operand order repaired and checked by reading, not compiled, not run on hardware
//
#include <hip/hip_runtime.h>

#define HN 50000
#define HE 800000

typedef __attribute__((ext_vector_type(16))) _Float16 v16h;
typedef __attribute__((ext_vector_type(8)))  _Float16 h8;
typedef __attribute__((ext_vector_type(8)))  float    v8f;

__device__ __forceinline__ float lrelu02(float x){ return x > 0.f ? x : 0.2f*x; }
__device__ __forceinline__ float gelu_ex(float x){ return 0.5f*x*(1.f + erff(x*0.7071067811865475f)); }

// exact float atomic max via sign-split int atomics
__device__ __forceinline__ void atomicMaxF(float* a, float v){
  if (v >= 0.f) atomicMax(reinterpret_cast<int*>(a), __float_as_int(v));
  else          atomicMin(reinterpret_cast<unsigned int*>(a), __float_as_uint(v));
}

// ---------------------------------------------------------------------------
// WMMA GEMM:  Y[N,M] = Xh[N,(ldx)] (f16, cols 0..K-1) @ Wt[M,K]^T (f16) + bias
// Block = 128 threads (4 waves). Block handles 16 rows; waves stride col tiles.
// A layout per CDNA5 ISA (16-bit A 16x32), B per 32x16 column layout, D per
// 8-VGPR C/D map.
// ---------------------------------------------------------------------------
__global__ void __launch_bounds__(128)
wmma_gemm_f16f32(const _Float16* __restrict__ Xh, int ldx,
                 const _Float16* __restrict__ Wt,
                 const float* __restrict__ bias,
                 float* __restrict__ Y,
                 int N, int K, int M)
{
  const int lane = threadIdx.x & 31;
  const int wave = threadIdx.x >> 5;
  const int half = lane >> 4;     // 0 or 1
  const int l16  = lane & 15;
  const int row0 = blockIdx.x << 4;

  int arow = row0 + l16;
  if (arow > N - 1) arow = N - 1;              // clamp (N%16==0 for our sizes)
  const _Float16* __restrict__ Arow = Xh + (size_t)arow * ldx;

  const int ntiles = M >> 4;
  for (int ct = wave; ct < ntiles; ct += 4) {
    const int col0 = ct << 4;
    const _Float16* __restrict__ Brow = Wt + (size_t)(col0 + l16) * K;
    v8f acc = {};
    for (int k0 = 0; k0 < K; k0 += 32) {
      // A: lanes<16 hold K {k0..k0+7, k0+16..k0+23}; lanes>=16 shifted by 8
      h8 alo = *(const h8*)(Arow + k0 + 8 * half);
      h8 ahi = *(const h8*)(Arow + k0 + 16 + 8 * half);
      v16h a = __builtin_shufflevector(alo, ahi, 0,1,2,3,4,5,6,7,8,9,10,11,12,13,14,15);
      // B: lane holds column col0+l16; lanes<16 K {k0..k0+7, k0+16..k0+23}; lanes>=16 shifted by 8
      h8 blo = *(const h8*)(Brow + k0 + 8 * half);
      h8 bhi = *(const h8*)(Brow + k0 + 16 + 8 * half);
      v16h b = __builtin_shufflevector(blo, bhi, 0,1,2,3,4,5,6,7,8,9,10,11,12,13,14,15);
      acc = __builtin_amdgcn_wmma_f32_16x16x32_f16(false, a, false, b, (short)0, acc, false, false);
    }
    const float bv = bias ? bias[col0 + l16] : 0.f;
    #pragma unroll
    for (int j = 0; j < 8; ++j) {
      const int r = row0 + j + 8 * half;       // D: VGPR j -> row j (+8 for upper half)
      if (r < N) Y[(size_t)r * M + col0 + l16] = acc[j] + bv;
    }
  }
}

// f32 -> f16 convert, with destination row stride / column offset (for concats)
__global__ void f32_to_f16_k(const float* __restrict__ src, _Float16* __restrict__ dst,
                             int N, int K, int stride, int off)
{
  long long t = (long long)blockIdx.x * blockDim.x + threadIdx.x;
  if (t >= (long long)N * K) return;
  int n = (int)(t / K), k = (int)(t % K);
  dst[(size_t)n * stride + off + k] = (_Float16)src[t];
}

// W[K,M] f32 -> Wt[M,K] f16 (transpose + convert)
__global__ void pack_weight_k(const float* __restrict__ W, _Float16* __restrict__ Wt,
                              int K, int M)
{
  int t = blockIdx.x * blockDim.x + threadIdx.x;
  if (t >= K * M) return;
  int m = t / K, k = t - m * K;
  Wt[t] = (_Float16)W[(size_t)k * M + m];
}

// in-place row LayerNorm + activation. mode: 0=none 1=relu 2=gelu. d in {64,128}
__global__ void __launch_bounds__(256)
ln_act_k(float* __restrict__ X, const float* __restrict__ g, const float* __restrict__ b,
         int N, int d, int mode)
{
  int row = blockIdx.x * (blockDim.x >> 5) + (threadIdx.x >> 5);
  if (row >= N) return;
  int lane = threadIdx.x & 31;
  float* xr = X + (size_t)row * d;
  int per = d >> 5;
  float vals[4];
  float s = 0.f;
  for (int i = 0; i < per; ++i) { vals[i] = xr[i * 32 + lane]; s += vals[i]; }
  for (int o = 16; o > 0; o >>= 1) s += __shfl_xor(s, o, 32);
  float mean = s / (float)d;
  float v = 0.f;
  for (int i = 0; i < per; ++i) { float t = vals[i] - mean; v += t * t; }
  for (int o = 16; o > 0; o >>= 1) v += __shfl_xor(v, o, 32);
  float inv = rsqrtf(v / (float)d + 1e-5f);
  for (int i = 0; i < per; ++i) {
    int c = i * 32 + lane;
    float y = (vals[i] - mean) * inv * g[c] + b[c];
    if (mode == 1) y = fmaxf(y, 0.f);
    else if (mode == 2) y = gelu_ex(y);
    xr[c] = y;
  }
}

// per (node, head): attention logit pieces; init m with self-loop logit, denom=0
__global__ void gat_node_prep(const float* __restrict__ h,
                              const float* __restrict__ aSrc, const float* __restrict__ aDst,
                              float* __restrict__ as_, float* __restrict__ ad_,
                              float* __restrict__ m, float* __restrict__ den,
                              int N, int H, int C)
{
  int t = blockIdx.x * blockDim.x + threadIdx.x;
  if (t >= N * H) return;
  int n = t / H, hd = t - n * H;
  const float* hr = h + (size_t)n * 128 + hd * C;
  const float* as = aSrc + hd * C;
  const float* ad = aDst + hd * C;
  float s = 0.f, d = 0.f;
  for (int c = 0; c < C; ++c) { s += hr[c] * as[c]; d += hr[c] * ad[c]; }
  as_[t] = s; ad_[t] = d;
  m[t] = lrelu02(s + d);   // self-loop logit (always present -> valid max init)
  den[t] = 0.f;
}

__global__ void gat_edge_max(const int* __restrict__ src, const int* __restrict__ dst,
                             const int* __restrict__ et, int rel,
                             const float* __restrict__ as_, const float* __restrict__ ad_,
                             float* __restrict__ m, int E, int H)
{
  int e = blockIdx.x * blockDim.x + threadIdx.x;
  if (e >= E) return;
  if (rel >= 0 && et[e] != rel) return;
  int s = src[e], d = dst[e];
  for (int hd = 0; hd < H; ++hd) {
    float v = lrelu02(as_[(size_t)s * H + hd] + ad_[(size_t)d * H + hd]);
    atomicMaxF(&m[(size_t)d * H + hd], v);
  }
}

__global__ void gat_edge_expsum(const int* __restrict__ src, const int* __restrict__ dst,
                                const int* __restrict__ et, int rel,
                                const float* __restrict__ as_, const float* __restrict__ ad_,
                                const float* __restrict__ m, float* __restrict__ den,
                                float* __restrict__ exb, const float* __restrict__ h,
                                int E, int H)
{
  int e = blockIdx.x * blockDim.x + threadIdx.x;
  if (e >= E) return;
  if (rel >= 0 && et[e] != rel) return;
  int s = src[e], d = dst[e];
  // warm L2 with the source row for the upcoming scatter pass
  __builtin_prefetch(h + (size_t)s * 128, 0, 1);
  for (int hd = 0; hd < H; ++hd) {
    float v = lrelu02(as_[(size_t)s * H + hd] + ad_[(size_t)d * H + hd]);
    float ex = expf(v - m[(size_t)d * H + hd]);
    exb[(size_t)e * H + hd] = ex;
    atomicAdd(&den[(size_t)d * H + hd], ex);
  }
}

__global__ void gat_selfdenom(const float* __restrict__ as_, const float* __restrict__ ad_,
                              const float* __restrict__ m, float* __restrict__ den, int NH)
{
  int t = blockIdx.x * blockDim.x + threadIdx.x;
  if (t >= NH) return;
  den[t] += expf(lrelu02(as_[t] + ad_[t]) - m[t]);
}

// one wave per edge: 4 passes, each 32 lanes x 1 float over 32 contiguous columns -> coalesced 512B scatter-add
__global__ void __launch_bounds__(256)
gat_edge_scatter(const int* __restrict__ src, const int* __restrict__ dst,
                 const int* __restrict__ et, int rel,
                 const float* __restrict__ h, const float* __restrict__ exb,
                 const float* __restrict__ den, float* __restrict__ out,
                 int E, int H, int Cshift)
{
  int e = blockIdx.x * (blockDim.x >> 5) + (threadIdx.x >> 5);
  if (e >= E) return;
  if (rel >= 0 && et[e] != rel) return;
  int s = src[e], d = dst[e];
  int lane = threadIdx.x & 31;
  #pragma unroll
  for (int i = 0; i < 4; ++i) {
    int c = i * 32 + lane;
    int hd = c >> Cshift;
    float alpha = exb[(size_t)e * H + hd] / den[(size_t)d * H + hd];
    atomicAdd(&out[(size_t)d * 128 + c], h[(size_t)s * 128 + c] * alpha);
  }
}

// add self-loop message + bias -> z
__global__ void gat_node_out(const float* __restrict__ h,
                             const float* __restrict__ as_, const float* __restrict__ ad_,
                             const float* __restrict__ m, const float* __restrict__ den,
                             const float* __restrict__ bias,
                             const float* __restrict__ acc, float* __restrict__ z,
                             int N, int H, int Cshift)
{
  int t = blockIdx.x * blockDim.x + threadIdx.x;
  if (t >= N * 128) return;
  int n = t >> 7, c = t & 127;
  int hd = c >> Cshift;
  float es = lrelu02(as_[(size_t)n * H + hd] + ad_[(size_t)n * H + hd]);
  float alpha = expf(es - m[(size_t)n * H + hd]) / den[(size_t)n * H + hd];
  z[t] = acc[t] + h[t] * alpha + bias[c];
}

// semantic: w[r,n] = sum_j tanh(tmp[n,j]) * w2[j]   (tmp already has +b1)
__global__ void __launch_bounds__(256)
tanh_dot_k(const float* __restrict__ tmp, const float* __restrict__ w2,
           float* __restrict__ wsem, int N, int r)
{
  int row = blockIdx.x * (blockDim.x >> 5) + (threadIdx.x >> 5);
  if (row >= N) return;
  int lane = threadIdx.x & 31;
  float s = 0.f;
  #pragma unroll
  for (int i = 0; i < 8; ++i) {
    int j = i * 32 + lane;
    s += tanhf(tmp[(size_t)row * 256 + j]) * w2[j];
  }
  for (int o = 16; o > 0; o >>= 1) s += __shfl_xor(s, o, 32);
  if (lane == 0) wsem[(size_t)r * N + row] = s;
}

// beta = softmax over R=3 of wsem; out = sum_r beta_r * z_r
__global__ void han_combine_k(const float* __restrict__ z0, const float* __restrict__ z1,
                              const float* __restrict__ z2, const float* __restrict__ wsem,
                              float* __restrict__ out, int N)
{
  int t = blockIdx.x * blockDim.x + threadIdx.x;
  if (t >= N * 128) return;
  int n = t >> 7;
  float w0 = wsem[n], w1 = wsem[N + n], w2 = wsem[2 * N + n];
  float mx = fmaxf(w0, fmaxf(w1, w2));
  float e0 = expf(w0 - mx), e1 = expf(w1 - mx), e2 = expf(w2 - mx);
  float inv = 1.f / (e0 + e1 + e2);
  out[t] = (e0 * z0[t] + e1 * z1[t] + e2 * z2[t]) * inv;
}

__global__ void residual_k(float* __restrict__ x3, const float* __restrict__ x1,
                           const float* __restrict__ x2, int total)
{
  int t = blockIdx.x * blockDim.x + threadIdx.x;
  if (t >= total) return;
  x3[t] = x3[t] + 0.2f * x1[t] + 0.3f * x2[t];
}

__global__ void log_softmax16_k(float* __restrict__ Y, int N)
{
  int n = blockIdx.x * blockDim.x + threadIdx.x;
  if (n >= N) return;
  float* r = Y + (size_t)n * 16;
  float v[16];
  float mx = -3.4e38f;
  #pragma unroll
  for (int i = 0; i < 16; ++i) { v[i] = r[i]; mx = fmaxf(mx, v[i]); }
  float s = 0.f;
  #pragma unroll
  for (int i = 0; i < 16; ++i) s += expf(v[i] - mx);
  float ls = mx + logf(s);
  #pragma unroll
  for (int i = 0; i < 16; ++i) r[i] = v[i] - ls;
}

// ---------------------------------------------------------------------------
// Host orchestration
// ---------------------------------------------------------------------------
extern "C" void kernel_launch(void* const* d_in, const int* in_sizes, int n_in,
                              void* d_out, int out_size, void* d_ws, size_t ws_size,
                              hipStream_t stream)
{
  (void)n_in; (void)out_size; (void)ws_size;
  const int N = HN, E = HE;

  auto F  = [&](int i) { return (const float*)d_in[i]; };
  auto Ii = [&](int i) { return (const int*)d_in[i]; };

  // Disambiguate leaf ordering: insertion-order has x (25.6M) first;
  // sorted-pytree order has edge_index (1.6M) first.
  const bool sortedOrder = (in_sizes[0] != N * 512);

  const float *x, *feW, *feB, *feG, *feLb;
  const int *ei, *et;
  const float *gW[4][3], *gAs[4][3], *gAd[4][3], *gB[4][3];
  const float *semW1[3], *semB1[3], *semw2[3];
  const float *lnG[3], *lnB[3], *lnAttG, *lnAttB;
  const float *fusW, *fusB, *fusG, *fusLb;
  const float *cW1, *cB1, *cG1, *cLb1, *cW2, *cB2, *cG2, *cLb2, *cW3, *cB3;

  if (!sortedOrder) {
    x = F(0); ei = Ii(1); et = Ii(2);
    feW = F(3); feB = F(4); feG = F(5); feLb = F(6);
    for (int L = 0; L < 3; ++L) {
      int base = 7 + L * 17;
      for (int r = 0; r < 3; ++r) {
        gW[L][r] = F(base + r * 4 + 0); gAs[L][r] = F(base + r * 4 + 1);
        gAd[L][r] = F(base + r * 4 + 2); gB[L][r] = F(base + r * 4 + 3);
      }
      semW1[L] = F(base + 12); semB1[L] = F(base + 13); semw2[L] = F(base + 14);
      lnG[L] = F(base + 15); lnB[L] = F(base + 16);
    }
    gW[3][0] = F(58); gAs[3][0] = F(59); gAd[3][0] = F(60); gB[3][0] = F(61);
    lnAttG = F(62); lnAttB = F(63);
    fusW = F(64); fusB = F(65); fusG = F(66); fusLb = F(67);
    cW1 = F(68); cB1 = F(69); cG1 = F(70); cLb1 = F(71);
    cW2 = F(72); cB2 = F(73); cG2 = F(74); cLb2 = F(75);
    cW3 = F(76); cB3 = F(77);
  } else {
    ei = Ii(0); et = Ii(1);
    cW1 = F(2); cW2 = F(3); cW3 = F(4); cB1 = F(5); cB2 = F(6); cB3 = F(7);
    cLb1 = F(8); cG1 = F(9); cLb2 = F(10); cG2 = F(11);
    feW = F(12); feB = F(13); feLb = F(14); feG = F(15);
    fusW = F(16); fusB = F(17); fusLb = F(18); fusG = F(19);
    gW[3][0] = F(20); gAd[3][0] = F(21); gAs[3][0] = F(22); gB[3][0] = F(23);
    for (int L = 0; L < 3; ++L) {
      int base = 24 + L * 15;
      for (int r = 0; r < 3; ++r) {
        gW[L][r] = F(base + r * 4 + 0); gAd[L][r] = F(base + r * 4 + 1);
        gAs[L][r] = F(base + r * 4 + 2); gB[L][r] = F(base + r * 4 + 3);
      }
      semW1[L] = F(base + 12); semB1[L] = F(base + 13); semw2[L] = F(base + 14);
    }
    lnB[0] = F(69); lnG[0] = F(70); lnB[1] = F(71); lnG[1] = F(72);
    lnB[2] = F(73); lnG[2] = F(74); lnAttB = F(75); lnAttG = F(76);
    x = F(77);
  }

  const int* src = ei;        // edge_index[0]
  const int* dst = ei + E;    // edge_index[1]

  // ---- workspace bump allocator ----
  char* wp = (char*)d_ws;
  auto alloc = [&](size_t bytes) { void* r = wp; wp += (bytes + 255) & ~(size_t)255; return r; };
  _Float16* XH    = (_Float16*)alloc((size_t)N * 512 * 2);
  _Float16* WT    = (_Float16*)alloc((size_t)512 * 128 * 2);
  float* x0    = (float*)alloc((size_t)N * 128 * 4);
  float* x1    = (float*)alloc((size_t)N * 128 * 4);
  float* x2    = (float*)alloc((size_t)N * 128 * 4);
  float* x3    = (float*)alloc((size_t)N * 128 * 4);
  float* z0    = (float*)alloc((size_t)N * 128 * 4);
  float* z1    = (float*)alloc((size_t)N * 128 * 4);
  float* z2    = (float*)alloc((size_t)N * 128 * 4);
  float* hbuf  = (float*)alloc((size_t)N * 128 * 4);
  float* outac = (float*)alloc((size_t)N * 128 * 4);
  float* xatt  = (float*)alloc((size_t)N * 128 * 4);
  float* fused = (float*)alloc((size_t)N * 128 * 4);
  float* hc1   = (float*)alloc((size_t)N * 128 * 4);
  float* tmp256= (float*)alloc((size_t)N * 256 * 4);
  float* hc2   = (float*)alloc((size_t)N * 64 * 4);
  float* asb   = (float*)alloc((size_t)N * 8 * 4);
  float* adb   = (float*)alloc((size_t)N * 8 * 4);
  float* mb    = (float*)alloc((size_t)N * 8 * 4);
  float* db    = (float*)alloc((size_t)N * 8 * 4);
  float* exb   = (float*)alloc((size_t)E * 8 * 4);
  float* wsem  = (float*)alloc((size_t)3 * N * 4);
  float* zbuf[3] = { z0, z1, z2 };

  const int TPB = 256;
  auto cdiv = [](long long a, long long b) { return (unsigned)((a + b - 1) / b); };

  auto conv = [&](const float* s, int K, int stride, int off) {
    f32_to_f16_k<<<cdiv((long long)N * K, TPB), TPB, 0, stream>>>(s, XH, N, K, stride, off);
  };
  auto gemm = [&](int ldx, const float* W, const float* bias, float* Y, int K, int M) {
    pack_weight_k<<<cdiv(K * M, TPB), TPB, 0, stream>>>(W, WT, K, M);
    wmma_gemm_f16f32<<<cdiv(N, 16), 128, 0, stream>>>(XH, ldx, WT, bias, Y, N, K, M);
  };
  auto lnact = [&](float* Y, const float* g, const float* b, int d, int mode) {
    ln_act_k<<<cdiv(N, 8), 256, 0, stream>>>(Y, g, b, N, d, mode);
  };
  auto run_gat = [&](const float* W, const float* aS, const float* aD, const float* bgat,
                     int rel, int H, float* zout) {
    const int C = 128 / H;
    const int Cshift = (C == 16) ? 4 : 5;
    gemm(128, W, nullptr, hbuf, 128, 128);                       // h = xh @ W
    gat_node_prep<<<cdiv(N * H, TPB), TPB, 0, stream>>>(hbuf, aS, aD, asb, adb, mb, db, N, H, C);
    hipMemsetAsync(outac, 0, (size_t)N * 128 * 4, stream);
    gat_edge_max<<<cdiv(E, TPB), TPB, 0, stream>>>(src, dst, et, rel, asb, adb, mb, E, H);
    gat_edge_expsum<<<cdiv(E, TPB), TPB, 0, stream>>>(src, dst, et, rel, asb, adb, mb, db, exb, hbuf, E, H);
    gat_selfdenom<<<cdiv(N * H, TPB), TPB, 0, stream>>>(asb, adb, mb, db, N * H);
    gat_edge_scatter<<<cdiv(E, 8), 256, 0, stream>>>(src, dst, et, rel, hbuf, exb, db, outac, E, H, Cshift);
    gat_node_out<<<cdiv((long long)N * 128, TPB), TPB, 0, stream>>>(hbuf, asb, adb, mb, db, bgat, outac, zout, N, H, Cshift);
  };

  // ---- feature encoder: x0 = gelu(ln(x @ feW + feB)) ----
  conv(x, 512, 512, 0);
  gemm(512, feW, feB, x0, 512, 128);
  lnact(x0, feG, feLb, 128, 2);

  // ---- 3 HAN layers ----
  const int Hs[3] = { 8, 4, 4 };
  float* xin = x0;
  float* xouts[3] = { x1, x2, x3 };
  for (int L = 0; L < 3; ++L) {
    conv(xin, 128, 128, 0);                                      // xh = f16(xin)
    for (int r = 0; r < 3; ++r)
      run_gat(gW[L][r], gAs[L][r], gAd[L][r], gB[L][r], r, Hs[L], zbuf[r]);
    for (int r = 0; r < 3; ++r) {                                // semantic attention
      conv(zbuf[r], 128, 128, 0);
      gemm(128, semW1[L], semB1[L], tmp256, 128, 256);
      tanh_dot_k<<<cdiv(N, 8), 256, 0, stream>>>(tmp256, semw2[L], wsem, N, r);
    }
    han_combine_k<<<cdiv((long long)N * 128, TPB), TPB, 0, stream>>>(z0, z1, z2, wsem, xouts[L], N);
    lnact(xouts[L], lnG[L], lnB[L], 128, 1);
    xin = xouts[L];
  }

  // ---- residual: x3 += 0.2*x1 + 0.3*x2 ----
  residual_k<<<cdiv((long long)N * 128, TPB), TPB, 0, stream>>>(x3, x1, x2, N * 128);

  // ---- final GAT over all edges + ln ----
  conv(x3, 128, 128, 0);
  run_gat(gW[3][0], gAs[3][0], gAd[3][0], gB[3][0], -1, 4, xatt);
  lnact(xatt, lnAttG, lnAttB, 128, 0);

  // ---- fusion: fused = gelu(ln(concat(x1,x2,x3) @ fusW + fusB)) ----
  conv(x1, 128, 384, 0); conv(x2, 128, 384, 128); conv(x3, 128, 384, 256);
  gemm(384, fusW, fusB, fused, 384, 128);
  lnact(fused, fusG, fusLb, 128, 2);

  // ---- classifier ----
  conv(fused, 128, 256, 0); conv(xatt, 128, 256, 128);
  gemm(256, cW1, cB1, hc1, 256, 128);
  lnact(hc1, cG1, cLb1, 128, 2);
  conv(hc1, 128, 128, 0);
  gemm(128, cW2, cB2, hc2, 128, 64);
  ln_act_k<<<cdiv(N, 8), 256, 0, stream>>>(hc2, cG2, cLb2, N, 64, 2);
  conv(hc2, 64, 64, 0);
  gemm(64, cW3, cB3, (float*)d_out, 64, 16);
  log_softmax16_k<<<cdiv(N, TPB), TPB, 0, stream>>>((float*)d_out, N);
}
